// RGATModel_15393162788998
// MI455X (gfx1250) — matrix operand order repaired and checked by reading, not compiled, not run on hardware
//
#include <hip/hip_runtime.h>

#define NODES   20000
#define EMB     128
#define HID     64
#define RELS    8
#define NCLS    16
#define NHEADS  3
#define NEDGES  640000
#define OUT1    (NHEADS * HID)    // 192
#define OUT2    (NHEADS * NCLS)   // 48
#define NEG_SLOPE 0.2f

typedef __attribute__((ext_vector_type(16))) _Float16 v16h;
typedef __attribute__((ext_vector_type(8)))  _Float16 v8h;
typedef __attribute__((ext_vector_type(8)))  float    v8f;

struct Frag { v8h lo, hi; };   // 32 B, bit-cast to v16h

// ---------------------------------------------------------------- utilities

__global__ void cvt_f16(const float* __restrict__ src, _Float16* __restrict__ dst, int n) {
    int i = blockIdx.x * blockDim.x + threadIdx.x;
    if (i < n) dst[i] = (_Float16)src[i];
}

__global__ void fill_u32(unsigned int* __restrict__ p, unsigned int v, int n) {
    int i = blockIdx.x * blockDim.x + threadIdx.x;
    if (i < n) p[i] = v;
}

// Float atomic max via sign-split (int max for +, uint min for -); init bits 0xFF800000 (-inf)
__device__ __forceinline__ void atomicMaxF(float* addr, float val) {
    unsigned int u = __float_as_uint(val);
    if (u & 0x80000000u) atomicMin((unsigned int*)addr, u);
    else                 atomicMax((int*)addr, (int)u);
}

// ---------------------------------------------------------------- weight pre-pack
// Convert f32 W[RELS,K,N] into f16 WMMA B-fragment order:
// Wp[((r*KT + kt)*NT + nt)*32 + lane][i] = W[r][kt*32 + (i>>3)*16 + (lane>>4)*8 + (i&7)][nt*16 + (lane&15)]
// so each lane's 16-half B fragment is one contiguous 32-byte chunk.
template<int K, int N>
__global__ void pack_w(const float* __restrict__ W, _Float16* __restrict__ Wp) {
    constexpr int KT = K / 32, NT = N / 16;
    int idx = blockIdx.x * blockDim.x + threadIdx.x;
    if (idx >= RELS * KT * NT * 32 * 16) return;
    int i    = idx & 15;
    int lane = (idx >> 4) & 31;
    int frag = idx >> 9;              // (r*KT + kt)*NT + nt
    int nt   = frag % NT;
    int kt   = (frag / NT) % KT;
    int r    = frag / (NT * KT);
    int k = kt * 32 + (i >> 3) * 16 + (lane >> 4) * 8 + (i & 7);
    int n = nt * 16 + (lane & 15);
    Wp[idx] = (_Float16)W[((size_t)r * K + k) * N + n];
}

// ---------------------------------------------------------------- WMMA GEMM
// XR[r,n,:] = X[n,:] @ W[r,:,:]   (f16 in, f32 accumulate, f16 out)
// One wave per 16x16 C tile. A fragment = two contiguous b128 loads per lane
// (ISA 7.12.2 16-bit A layout); B fragment = one contiguous 32B chunk of
// pre-packed W. NODES is a multiple of 16 -> no partial tiles, no guards.
template<int K, int N>
__global__ __launch_bounds__(32)
void gemm_wmma(const _Float16* __restrict__ X,    // [NODES, K]
               const _Float16* __restrict__ Wp,   // packed fragments
               _Float16* __restrict__ XR)         // [RELS, NODES, N]
{
    constexpr int KT = K / 32, NT = N / 16;
    const int lane = threadIdx.x & 31;
    const int g  = lane >> 4;
    const int lr = lane & 15;
    const int nt = blockIdx.x;
    const int m0 = blockIdx.y * 16;
    const int r  = blockIdx.z;

    v8f c = {0.f, 0.f, 0.f, 0.f, 0.f, 0.f, 0.f, 0.f};

    const _Float16* Xrow  = X + (size_t)(m0 + lr) * K;
    const _Float16* Wbase = Wp + ((((size_t)r * KT) * NT + nt) * 32 + lane) * 16;

#pragma unroll
    for (int kt = 0; kt < KT; ++kt) {
        Frag a, b;
        a.lo = *(const v8h*)(Xrow + kt * 32 + 8 * g);        // K = 8g .. 8g+7
        a.hi = *(const v8h*)(Xrow + kt * 32 + 16 + 8 * g);   // K = 16+8g .. 16+8g+7
        const _Float16* wf = Wbase + (size_t)kt * NT * 32 * 16;
        b.lo = *(const v8h*)(wf);
        b.hi = *(const v8h*)(wf + 8);
        c = __builtin_amdgcn_wmma_f32_16x16x32_f16(
                false, __builtin_bit_cast(v16h, a),
                false, __builtin_bit_cast(v16h, b),
                (short)0, c, false, false);
    }

    // C layout: VGPR i -> row M = i + 8*g, col N = lr
    _Float16* dst = XR + ((size_t)r * NODES + m0 + g * 8) * N + nt * 16 + lr;
#pragma unroll
    for (int i = 0; i < 8; ++i)
        dst[(size_t)i * N] = (_Float16)c[i];
}

// ---------------------------------------------------------------- attention scalars
// nq[r,n,h] = xr[r,n,:] . Q[:,h] ;  nk likewise  (factored: avoids [E,OUT] gathers for logits)
template<int OUT>
__global__ void node_qk(const _Float16* __restrict__ XR,
                        const float* __restrict__ Q,   // [OUT, NHEADS]
                        const float* __restrict__ Km,  // [OUT, NHEADS]
                        float* __restrict__ nq, float* __restrict__ nk)
{
    int idx = blockIdx.x * blockDim.x + threadIdx.x;   // r*NODES + n
    if (idx >= RELS * NODES) return;
    const _Float16* x = XR + (size_t)idx * OUT;
    float aq[NHEADS] = {0.f, 0.f, 0.f}, ak[NHEADS] = {0.f, 0.f, 0.f};
    for (int o = 0; o < OUT; ++o) {
        float v = (float)x[o];
#pragma unroll
        for (int h = 0; h < NHEADS; ++h) {
            aq[h] += v * Q[o * NHEADS + h];
            ak[h] += v * Km[o * NHEADS + h];
        }
    }
#pragma unroll
    for (int h = 0; h < NHEADS; ++h) {
        nq[(size_t)idx * NHEADS + h] = aq[h];
        nk[(size_t)idx * NHEADS + h] = ak[h];
    }
}

// ---------------------------------------------------------------- edge passes
__global__ void edge_alpha(const int* __restrict__ ei, const int* __restrict__ et,
                           const float* __restrict__ nq, const float* __restrict__ nk,
                           float* __restrict__ aE, float* __restrict__ amax)
{
    int e = blockIdx.x * blockDim.x + threadIdx.x;
    if (e >= NEDGES) return;
    int s = ei[e], d = ei[NEDGES + e], r = et[e];
    size_t bi = ((size_t)r * NODES + d) * NHEADS;
    size_t bj = ((size_t)r * NODES + s) * NHEADS;
#pragma unroll
    for (int h = 0; h < NHEADS; ++h) {
        float a = nq[bi + h] + nk[bj + h];
        a = a > 0.f ? a : NEG_SLOPE * a;           // leaky_relu
        aE[(size_t)e * NHEADS + h] = a;
        atomicMaxF(&amax[d * NHEADS + h], a);
    }
}

__global__ void edge_exp(const int* __restrict__ ei, float* __restrict__ aE,
                         const float* __restrict__ amax, float* __restrict__ denom)
{
    int e = blockIdx.x * blockDim.x + threadIdx.x;
    if (e >= NEDGES) return;
    int d = ei[NEDGES + e];
#pragma unroll
    for (int h = 0; h < NHEADS; ++h) {
        float ex = __expf(aE[(size_t)e * NHEADS + h] - amax[d * NHEADS + h]);
        aE[(size_t)e * NHEADS + h] = ex;           // overwrite in place
        atomicAdd(&denom[d * NHEADS + h], ex);
    }
}

template<int OUT, int HD>
__global__ void edge_aggr(const int* __restrict__ ei, const int* __restrict__ et,
                          const float* __restrict__ exE, const float* __restrict__ denom,
                          const _Float16* __restrict__ XR, float* __restrict__ aggr)
{
    long long idx = (long long)blockIdx.x * blockDim.x + threadIdx.x;
    if (idx >= (long long)NEDGES * OUT) return;
    int e = (int)(idx / OUT);
    int o = (int)(idx % OUT);
    int h = o / HD;
    int s = ei[e], d = ei[NEDGES + e], r = et[e];
    float alpha = exE[(size_t)e * NHEADS + h] / (denom[d * NHEADS + h] + 1e-16f);
    float v = (float)XR[((size_t)r * NODES + s) * OUT + o];
    atomicAdd(&aggr[(size_t)d * OUT + o], alpha * v);
}

// ---------------------------------------------------------------- finalize
__global__ void finalize1(const float* __restrict__ aggr, const float* __restrict__ b,
                          _Float16* __restrict__ x2h)
{
    int idx = blockIdx.x * blockDim.x + threadIdx.x;
    if (idx >= NODES * HID) return;
    int n = idx / HID, o = idx % HID;
    const float* a = aggr + (size_t)n * OUT1;
    float m = (a[o] + a[HID + o] + a[2 * HID + o]) * (1.f / 3.f) + b[o];
    x2h[idx] = (_Float16)(m > 0.f ? m : 0.f);      // relu, f16 for next GEMM
}

__global__ void finalize2(const float* __restrict__ aggr, const float* __restrict__ b,
                          float* __restrict__ out)
{
    int idx = blockIdx.x * blockDim.x + threadIdx.x;
    if (idx >= NODES * NCLS) return;
    int n = idx / NCLS, c = idx % NCLS;
    const float* a = aggr + (size_t)n * OUT2;
    float m = (a[c] + a[NCLS + c] + a[2 * NCLS + c]) * (1.f / 3.f) + b[c];
    out[idx] = 1.f / (1.f + __expf(-m));           // sigmoid
}

// ---------------------------------------------------------------- launch
extern "C" void kernel_launch(void* const* d_in, const int* in_sizes, int n_in,
                              void* d_out, int out_size, void* d_ws, size_t ws_size,
                              hipStream_t stream)
{
    (void)in_sizes; (void)n_in; (void)out_size; (void)ws_size;
    const float* emb = (const float*)d_in[0];
    const float* w1  = (const float*)d_in[1];
    const float* q1  = (const float*)d_in[2];
    const float* k1  = (const float*)d_in[3];
    const float* b1  = (const float*)d_in[4];
    const float* w2  = (const float*)d_in[5];
    const float* q2  = (const float*)d_in[6];
    const float* k2  = (const float*)d_in[7];
    const float* b2  = (const float*)d_in[8];
    const int*   ei  = (const int*)d_in[9];   // [2, E]: row0=src, row1=dst
    const int*   et  = (const int*)d_in[10];  // [E]
    float* out = (float*)d_out;

    constexpr int W1P = RELS * (EMB / 32) * (OUT1 / 16) * 32 * 16;  // packed halves
    constexpr int W2P = RELS * (HID / 32) * (OUT2 / 16) * 32 * 16;

    // scratch carve-out (~97 MB peak; layer-2 reuses layer-1 buffers)
    char* p = (char*)d_ws;
    auto carve = [&](size_t bytes) -> char* {
        char* r = p; p += (bytes + 255) & ~(size_t)255; return r;
    };
    _Float16* xh    = (_Float16*)carve((size_t)NODES * EMB * 2);
    _Float16* w1p   = (_Float16*)carve((size_t)W1P * 2);
    _Float16* w2p   = (_Float16*)carve((size_t)W2P * 2);
    _Float16* xr    = (_Float16*)carve((size_t)RELS * NODES * OUT1 * 2); // reused for layer 2
    float*    nq    = (float*)carve((size_t)RELS * NODES * NHEADS * 4);
    float*    nk    = (float*)carve((size_t)RELS * NODES * NHEADS * 4);
    float*    aE    = (float*)carve((size_t)NEDGES * NHEADS * 4);        // logits then exp
    float*    amax  = (float*)carve((size_t)NODES * NHEADS * 4);
    float*    denom = (float*)carve((size_t)NODES * NHEADS * 4);
    float*    aggr  = (float*)carve((size_t)NODES * OUT1 * 4);           // reused for layer 2
    _Float16* x2h   = (_Float16*)carve((size_t)NODES * HID * 2);

    const int B = 256;
    auto blks = [](long long n, int b) { return (unsigned)((n + b - 1) / b); };

    // operand conversion / weight fragment packing
    cvt_f16<<<blks((long long)NODES * EMB, B), B, 0, stream>>>(emb, xh, NODES * EMB);
    pack_w<EMB, OUT1><<<blks(W1P, B), B, 0, stream>>>(w1, w1p);
    pack_w<HID, OUT2><<<blks(W2P, B), B, 0, stream>>>(w2, w2p);

    // ---------------- layer 1 ----------------
    gemm_wmma<EMB, OUT1><<<dim3(OUT1 / 16, NODES / 16, RELS), 32, 0, stream>>>(xh, w1p, xr);
    node_qk<OUT1><<<blks((long long)RELS * NODES, B), B, 0, stream>>>(xr, q1, k1, nq, nk);

    fill_u32<<<blks(NODES * NHEADS, B), B, 0, stream>>>((unsigned int*)amax, 0xFF800000u, NODES * NHEADS);
    fill_u32<<<blks(NODES * NHEADS, B), B, 0, stream>>>((unsigned int*)denom, 0u, NODES * NHEADS);
    fill_u32<<<blks((long long)NODES * OUT1, B), B, 0, stream>>>((unsigned int*)aggr, 0u, NODES * OUT1);

    edge_alpha<<<blks(NEDGES, B), B, 0, stream>>>(ei, et, nq, nk, aE, amax);
    edge_exp<<<blks(NEDGES, B), B, 0, stream>>>(ei, aE, amax, denom);
    edge_aggr<OUT1, HID><<<blks((long long)NEDGES * OUT1, B), B, 0, stream>>>(ei, et, aE, denom, xr, aggr);
    finalize1<<<blks(NODES * HID, B), B, 0, stream>>>(aggr, b1, x2h);

    // ---------------- layer 2 ----------------
    gemm_wmma<HID, OUT2><<<dim3(OUT2 / 16, NODES / 16, RELS), 32, 0, stream>>>(x2h, w2p, xr);
    node_qk<OUT2><<<blks((long long)RELS * NODES, B), B, 0, stream>>>(xr, q2, k2, nq, nk);

    fill_u32<<<blks(NODES * NHEADS, B), B, 0, stream>>>((unsigned int*)amax, 0xFF800000u, NODES * NHEADS);
    fill_u32<<<blks(NODES * NHEADS, B), B, 0, stream>>>((unsigned int*)denom, 0u, NODES * NHEADS);
    fill_u32<<<blks((long long)NODES * OUT2, B), B, 0, stream>>>((unsigned int*)aggr, 0u, NODES * OUT2);

    edge_alpha<<<blks(NEDGES, B), B, 0, stream>>>(ei, et, nq, nk, aE, amax);
    edge_exp<<<blks(NEDGES, B), B, 0, stream>>>(ei, aE, amax, denom);
    edge_aggr<OUT2, NCLS><<<blks((long long)NEDGES * OUT2, B), B, 0, stream>>>(ei, et, aE, denom, xr, aggr);
    finalize2<<<blks(NODES * NCLS, B), B, 0, stream>>>(aggr, b2, out);
}
